// MultiheadSelfAttention_87686052315944
// MI455X (gfx1250) — matrix operand order repaired and checked by reading, not compiled, not run on hardware
//
#include <hip/hip_runtime.h>

typedef __bf16 bf16;
typedef __bf16 v4bf  __attribute__((ext_vector_type(4)));
typedef __bf16 v8bf  __attribute__((ext_vector_type(8)));
typedef __bf16 v16bf __attribute__((ext_vector_type(16)));
typedef float  v8f   __attribute__((ext_vector_type(8)));

#define DM   1024
#define NH   16
#define HD   64
#define B_   2
#define S_   2048
#define MT   (B_ * S_)   // 4096 total rows

// Use CDNA5 async global->LDS DMA when the toolchain exposes it.
#if defined(__AMDGCN__) && \
    __has_builtin(__builtin_amdgcn_global_load_async_to_lds_b128) && \
    __has_builtin(__builtin_amdgcn_s_wait_asynccnt)
#define USE_ASYNC 1
#else
#define USE_ASYNC 0
#endif

#if USE_ASYNC
typedef int gv4i __attribute__((vector_size(16)));
typedef __attribute__((address_space(1))) gv4i* gas1_v4i;
typedef __attribute__((address_space(3))) gv4i* gas3_v4i;

__device__ __forceinline__ void async_cp16(const bf16* g, bf16* l) {
  // global_load_async_to_lds_b128: 16 bytes / lane, memory -> LDS, ASYNCcnt.
  __builtin_amdgcn_global_load_async_to_lds_b128(
      (gas1_v4i)(g), (gas3_v4i)(l), 0, 0);
}
#endif

// ---- WMMA fragment loaders (wave32, v_wmma_f32_16x16x32_bf16) ----
// A fragment: lane holds row m = lane&15; 16 contiguous K-elements
// at {2*kb2..2*kb2+15}, kb2 = 8*(lane>>4) (matches ldB16 order).
__device__ __forceinline__ v16bf ldA(const bf16* p, int kb2) {
  v8bf lo = *reinterpret_cast<const v8bf*>(p + 2 * kb2);
  v8bf hi = *reinterpret_cast<const v8bf*>(p + 2 * kb2 + 8);
  v16bf r;
#pragma unroll
  for (int i = 0; i < 8; ++i) { r[i] = lo[i]; r[i + 8] = hi[i]; }
  return r;
}

// B fragment: lane holds column n = lane&15; 16 contiguous K-elements starting
// at 16*(lane>>4).  Caller passes pointer to element (n, 16*half).
__device__ __forceinline__ v16bf ldB16(const bf16* p) {
  v8bf lo = *reinterpret_cast<const v8bf*>(p);
  v8bf hi = *reinterpret_cast<const v8bf*>(p + 8);
  v16bf r;
#pragma unroll
  for (int i = 0; i < 8; ++i) { r[i] = lo[i]; r[i + 8] = hi[i]; }
  return r;
}

__device__ __forceinline__ v8f wmma_bf16(v16bf a, v16bf b, v8f c) {
  return __builtin_amdgcn_wmma_f32_16x16x32_bf16(false, a, false, b, (short)0, c,
                                                 false, false);
}

// ---- fp32 -> bf16 bulk convert (4 elements / thread) ----
__global__ __launch_bounds__(256) void cvt_bf16(const float* __restrict__ in,
                                                bf16* __restrict__ out, int n4) {
  int i = blockIdx.x * blockDim.x + threadIdx.x;
  if (i < n4) {
    const float4 f = reinterpret_cast<const float4*>(in)[i];
    v4bf o;
    o[0] = (bf16)f.x; o[1] = (bf16)f.y; o[2] = (bf16)f.z; o[3] = (bf16)f.w;
    reinterpret_cast<v4bf*>(out)[i] = o;
  }
}

// ---- C(f32, MxN) = A(bf16, MxK) @ W(bf16, NxK)^T ----
// 256 threads = 8 waves (4x2), block tile 128x128, K step 32,
// double-buffered LDS staged via async global->LDS copies.
__global__ __launch_bounds__(256) void gemm_bf16(const bf16* __restrict__ A,
                                                 const bf16* __restrict__ W,
                                                 float* __restrict__ C,
                                                 int M, int N, int K) {
  __shared__ alignas(16) bf16 As[2][128][32];
  __shared__ alignas(16) bf16 Ws[2][128][32];
  const int tid = threadIdx.x;
  const int lane = tid & 31, wid = tid >> 5;
  const int wm = wid >> 1, wn = wid & 1;            // 4 x 2 wave grid
  const int m0 = blockIdx.x * 128, n0 = blockIdx.y * 128;
  const int mlane = lane & 15, hl = lane >> 4, kb2 = hl * 8;
  const int lrow = tid >> 2, lcol = (tid & 3) * 8;  // staging: 64 rows/pass

  const bf16* Ar0 = &A[(size_t)(m0 + lrow) * K + lcol];
  const bf16* Ar1 = &A[(size_t)(m0 + lrow + 64) * K + lcol];
  const bf16* Wr0 = &W[(size_t)(n0 + lrow) * K + lcol];
  const bf16* Wr1 = &W[(size_t)(n0 + lrow + 64) * K + lcol];

  auto stage = [&](int bi, int kk) {
#if USE_ASYNC
    async_cp16(Ar0 + kk, &As[bi][lrow][lcol]);
    async_cp16(Ar1 + kk, &As[bi][lrow + 64][lcol]);
    async_cp16(Wr0 + kk, &Ws[bi][lrow][lcol]);
    async_cp16(Wr1 + kk, &Ws[bi][lrow + 64][lcol]);
#else
    *reinterpret_cast<v8bf*>(&As[bi][lrow][lcol]) =
        *reinterpret_cast<const v8bf*>(Ar0 + kk);
    *reinterpret_cast<v8bf*>(&As[bi][lrow + 64][lcol]) =
        *reinterpret_cast<const v8bf*>(Ar1 + kk);
    *reinterpret_cast<v8bf*>(&Ws[bi][lrow][lcol]) =
        *reinterpret_cast<const v8bf*>(Wr0 + kk);
    *reinterpret_cast<v8bf*>(&Ws[bi][lrow + 64][lcol]) =
        *reinterpret_cast<const v8bf*>(Wr1 + kk);
#endif
  };

  v8f acc[2][4];
#pragma unroll
  for (int a = 0; a < 2; ++a)
#pragma unroll
    for (int b = 0; b < 4; ++b)
#pragma unroll
      for (int r = 0; r < 8; ++r) acc[a][b][r] = 0.0f;

  stage(0, 0);                                      // prologue: buffer 0
  for (int kk = 0; kk < K; kk += 32) {
    const int cur = (kk >> 5) & 1;
    const int kn = kk + 32;
    stage(cur ^ 1, kn < K ? kn : 0);  // always issue; wrap copy is discarded
#if USE_ASYNC
    // async loads complete in order: <=4 outstanding => current tile landed
    __builtin_amdgcn_s_wait_asynccnt(4);
#endif
    __syncthreads();

    v16bf wfr[4];
#pragma unroll
    for (int in = 0; in < 4; ++in)
      wfr[in] = ldB16(&Ws[cur][wn * 64 + in * 16 + mlane][hl * 16]);
#pragma unroll
    for (int im = 0; im < 2; ++im) {
      v16bf af = ldA(&As[cur][wm * 32 + im * 16 + mlane][0], kb2);
#pragma unroll
      for (int in = 0; in < 4; ++in)
        acc[im][in] = wmma_bf16(af, wfr[in], acc[im][in]);
    }
    __syncthreads();   // all waves done with 'cur' before it is restaged
  }

#pragma unroll
  for (int im = 0; im < 2; ++im)
#pragma unroll
    for (int in = 0; in < 4; ++in)
#pragma unroll
      for (int r = 0; r < 8; ++r) {
        int row = m0 + wm * 32 + im * 16 + r + 8 * hl;
        int col = n0 + wn * 64 + in * 16 + mlane;
        C[(size_t)row * N + col] = acc[im][in][r];
      }
}

// ---- RoPE(Q,K) + bf16 pack of Q,K,V into (B,H,S,hd) layout ----
// one thread per (b,s,h,pair i), i in [0,32)
__global__ __launch_bounds__(256) void rope_pack(const float* __restrict__ Qf,
                                                 const float* __restrict__ Kf,
                                                 const float* __restrict__ Vf,
                                                 const int* __restrict__ pos,
                                                 bf16* __restrict__ Qb,
                                                 bf16* __restrict__ Kb,
                                                 bf16* __restrict__ Vb) {
  int t = blockIdx.x * blockDim.x + threadIdx.x;   // 2^21 threads
  int i = t & 31;
  int h = (t >> 5) & (NH - 1);
  int s = (t >> 9) & (S_ - 1);
  int b = t >> 20;
  float p = (float)pos[b * S_ + s];
  float ang = p * __powf(10000.0f, -(float)(2 * i) / (float)HD);
  float sn, cs;
  __sincosf(ang, &sn, &cs);
  size_t src = (size_t)(b * S_ + s) * DM + h * HD + 2 * i;
  size_t dst = ((size_t)(b * NH + h) * S_ + s) * HD + 2 * i;
  float q1 = Qf[src], q2 = Qf[src + 1];
  float k1 = Kf[src], k2 = Kf[src + 1];
  Qb[dst]     = (bf16)(q1 * cs - q2 * sn);
  Qb[dst + 1] = (bf16)(q1 * sn + q2 * cs);
  Kb[dst]     = (bf16)(k1 * cs - k2 * sn);
  Kb[dst + 1] = (bf16)(k1 * sn + k2 * cs);
  Vb[dst]     = (bf16)Vf[src];
  Vb[dst + 1] = (bf16)Vf[src + 1];
}

// ---- flash attention: block = 4 waves, each wave owns 16 q rows (64/block) ----
__global__ __launch_bounds__(128) void flash_attn(const bf16* __restrict__ Q,
                                                  const bf16* __restrict__ K,
                                                  const bf16* __restrict__ V,
                                                  bf16* __restrict__ O) {
  __shared__ alignas(16) bf16 Kt[32][HD];        // [key][d]
  __shared__ alignas(16) bf16 Vt[HD][32];        // [d][key] (transposed)
  __shared__ alignas(16) bf16 Pl[4][16][32];     // per-wave P staging
  const int lane = threadIdx.x & 31, w = threadIdx.x >> 5;
  const int mlane = lane & 15, hl = lane >> 4, kb2 = hl * 8;
  const int b = blockIdx.z, h = blockIdx.y;
  const int q0 = blockIdx.x * 64 + w * 16;
  const bf16* Qh = Q + (size_t)(b * NH + h) * S_ * HD;
  const bf16* Kh = K + (size_t)(b * NH + h) * S_ * HD;
  const bf16* Vh = V + (size_t)(b * NH + h) * S_ * HD;

  v16bf qf[2];
  {
    const bf16* qr = Qh + (size_t)(q0 + mlane) * HD;
    qf[0] = ldA(qr, kb2);        // d = 0..31
    qf[1] = ldA(qr + 32, kb2);   // d = 32..63
  }
  v8f o[4];
  float Mx[8], Ls[8];
#pragma unroll
  for (int r = 0; r < 8; ++r) { Mx[r] = -1e30f; Ls[r] = 0.0f; }
#pragma unroll
  for (int t = 0; t < 4; ++t)
#pragma unroll
    for (int r = 0; r < 8; ++r) o[t][r] = 0.0f;

  const int kend = blockIdx.x * 64 + 64;          // causal bound (uniform)
  for (int kb = 0; kb < kend; kb += 32) {
    // stage K (async DMA to LDS) and V (VGPR transpose): 2 x 8 bf16 per thread
#pragma unroll
    for (int j = 0; j < 2; ++j) {
      int c = threadIdx.x + 128 * j;
      int rr = c >> 3, col = (c & 7) * 8;
#if USE_ASYNC
      async_cp16(&Kh[(size_t)(kb + rr) * HD + col], &Kt[rr][col]);
#else
      *reinterpret_cast<v8bf*>(&Kt[rr][col]) =
          *reinterpret_cast<const v8bf*>(&Kh[(size_t)(kb + rr) * HD + col]);
#endif
      v8bf vv = *reinterpret_cast<const v8bf*>(&Vh[(size_t)(kb + rr) * HD + col]);
#pragma unroll
      for (int e = 0; e < 8; ++e) Vt[col + e][rr] = vv[e];
    }
#if USE_ASYNC
    __builtin_amdgcn_s_wait_asynccnt(0);
#endif
    __syncthreads();

    // S = Q @ K^T  (two 16-key tiles, K-dim = hd in 2 steps of 32)
    v8f sc[2];
#pragma unroll
    for (int t = 0; t < 2; ++t) {
#pragma unroll
      for (int r = 0; r < 8; ++r) sc[t][r] = 0.0f;
#pragma unroll
      for (int dk = 0; dk < 2; ++dk) {
        v16bf kf = ldB16(&Kt[t * 16 + mlane][dk * 32 + hl * 16]);
        sc[t] = wmma_bf16(qf[dk], kf, sc[t]);
      }
    }

    // online softmax (rows live across 16-lane groups; reduce via shfl_xor)
    float p0[8], p1[8];
#pragma unroll
    for (int r = 0; r < 8; ++r) {
      const int qr = q0 + r + 8 * hl;
      float s0 = sc[0][r] * 0.125f;                 // 1/sqrt(64)
      float s1 = sc[1][r] * 0.125f;
      if (kb + mlane > qr)      s0 = -1e30f;        // causal mask
      if (kb + 16 + mlane > qr) s1 = -1e30f;
      float mx = fmaxf(s0, s1);
#pragma unroll
      for (int d = 1; d < 16; d <<= 1) mx = fmaxf(mx, __shfl_xor(mx, d, 16));
      float Mn = fmaxf(Mx[r], mx);
      float al = __expf(Mx[r] - Mn);
      Mx[r] = Mn;
      float e0 = __expf(s0 - Mn);
      float e1 = __expf(s1 - Mn);
      float rs = e0 + e1;
#pragma unroll
      for (int d = 1; d < 16; d <<= 1) rs += __shfl_xor(rs, d, 16);
      Ls[r] = Ls[r] * al + rs;
#pragma unroll
      for (int t = 0; t < 4; ++t) o[t][r] = o[t][r] * al;
      p0[r] = e0; p1[r] = e1;
    }

    // reformat P (C layout) -> A fragment via per-wave LDS round trip
#pragma unroll
    for (int r = 0; r < 8; ++r) {
      int m = r + 8 * hl;
      Pl[w][m][mlane]      = (bf16)p0[r];
      Pl[w][m][16 + mlane] = (bf16)p1[r];
    }
    __builtin_amdgcn_wave_barrier();   // keep DS store->load order (same wave)
    v16bf pf = ldA(&Pl[w][mlane][0], kb2);
#pragma unroll
    for (int t = 0; t < 4; ++t) {
      v16bf vf = ldB16(&Vt[t * 16 + mlane][hl * 16]);
      o[t] = wmma_bf16(pf, vf, o[t]);
    }
    __syncthreads();
  }

  // epilogue: normalize, write O in (B,S,D) layout for the final projection
#pragma unroll
  for (int t = 0; t < 4; ++t)
#pragma unroll
    for (int r = 0; r < 8; ++r) {
      int qr = q0 + r + 8 * hl;
      float val = o[t][r] / Ls[r];
      O[(size_t)(b * S_ + qr) * DM + h * HD + t * 16 + mlane] = (bf16)val;
    }
}

extern "C" void kernel_launch(void* const* d_in, const int* in_sizes, int n_in,
                              void* d_out, int out_size, void* d_ws, size_t ws_size,
                              hipStream_t stream) {
  const float* x  = (const float*)d_in[0];
  const int*   tp = (const int*)d_in[1];
  const float* wq = (const float*)d_in[2];
  const float* wk = (const float*)d_in[3];
  const float* wv = (const float*)d_in[4];
  const float* wo = (const float*)d_in[5];
  char* ws = (char*)d_ws;
  const size_t MB = 1024 * 1024;

  bf16*  xb  = (bf16*)(ws + 0);         //  8 MB  x in bf16
  bf16*  wqb = (bf16*)(ws + 8  * MB);   //  2 MB
  bf16*  wkb = (bf16*)(ws + 10 * MB);   //  2 MB
  bf16*  wvb = (bf16*)(ws + 12 * MB);   //  2 MB
  bf16*  wob = (bf16*)(ws + 14 * MB);   //  2 MB
  float* Qf  = (float*)(ws + 16 * MB);  // 16 MB  (dead after rope_pack)
  float* Kf  = (float*)(ws + 32 * MB);  // 16 MB
  float* Vf  = (float*)(ws + 48 * MB);  // 16 MB
  bf16*  Qb  = (bf16*)(ws + 64 * MB);   //  8 MB  (B,H,S,hd)
  bf16*  Kb  = (bf16*)(ws + 72 * MB);   //  8 MB
  bf16*  Vb  = (bf16*)(ws + 80 * MB);   //  8 MB
  bf16*  Ob  = (bf16*)(ws + 16 * MB);   //  8 MB  aliases Qf (safe: Qf dead)

  cvt_bf16<<<(MT * DM / 4 + 255) / 256, 256, 0, stream>>>(x,  xb,  MT * DM / 4);
  cvt_bf16<<<(DM * DM / 4 + 255) / 256, 256, 0, stream>>>(wq, wqb, DM * DM / 4);
  cvt_bf16<<<(DM * DM / 4 + 255) / 256, 256, 0, stream>>>(wk, wkb, DM * DM / 4);
  cvt_bf16<<<(DM * DM / 4 + 255) / 256, 256, 0, stream>>>(wv, wvb, DM * DM / 4);
  cvt_bf16<<<(DM * DM / 4 + 255) / 256, 256, 0, stream>>>(wo, wob, DM * DM / 4);

  dim3 gp(MT / 128, DM / 128);
  gemm_bf16<<<gp, 256, 0, stream>>>(xb, wqb, Qf, MT, DM, DM);
  gemm_bf16<<<gp, 256, 0, stream>>>(xb, wkb, Kf, MT, DM, DM);
  gemm_bf16<<<gp, 256, 0, stream>>>(xb, wvb, Vf, MT, DM, DM);

  int nrope = B_ * S_ * NH * 32;   // 2^21
  rope_pack<<<nrope / 256, 256, 0, stream>>>(Qf, Kf, Vf, tp, Qb, Kb, Vb);

  flash_attn<<<dim3(S_ / 64, NH, B_), 128, 0, stream>>>(Qb, Kb, Vb, Ob);

  gemm_bf16<<<gp, 256, 0, stream>>>(Ob, wob, (float*)d_out, MT, DM, DM);
}
